// ConvAttention_21887153340981
// MI455X (gfx1250) — matrix operand order repaired and checked by reading, not compiled, not run on hardware
//
#include <hip/hip_runtime.h>

#define DEVINL __device__ __forceinline__

typedef __attribute__((ext_vector_type(16))) _Float16 v16h;
typedef __attribute__((ext_vector_type(8)))  _Float16 v8h;
typedef __attribute__((ext_vector_type(8)))  float    v8f;
typedef __attribute__((ext_vector_type(4)))  float    v4f;

constexpr int BN = 4;
constexpr int TN = 4096;
constexpr int CN = 256;
constexpr float QSCALE = 0.0625f;   // C^-0.5 = 1/16
constexpr float LN_EPS = 1e-5f;
constexpr int LDG = 264;            // padded LDS stride (halves) for 64x256 tiles
constexpr int LDP = 72;             // padded LDS stride (halves) for P^T tile

// ---------------------------------------------------------------- WMMA core
DEVINL v8f wmma_f16(v16h a, v16h b, v8f c) {
  return __builtin_amdgcn_wmma_f32_16x16x32_f16(
      /*neg_a=*/false, a, /*neg_b=*/false, b,
      /*c_mod=*/(short)0, c, /*reuse_a=*/false, /*reuse_b=*/false);
}

// A-fragment: 16x32 (MxK), row-major source with leading dim ld (elements).
DEVINL v16h load_a_frag(const _Float16* base, int ld) {
  const int lane = threadIdx.x & 31;
  const int m = lane & 15;
  const int g = lane >> 4;
  const _Float16* p = base + (size_t)m * ld + g * 8;
  v8h lo = *(const v8h*)p;
  v8h hi = *(const v8h*)(p + 16);
  v16h a;
#pragma unroll
  for (int i = 0; i < 8; ++i) { a[i] = lo[i]; a[i + 8] = hi[i]; }
  return a;
}

// B-fragment: 32x16 (KxN), element (k,n) at base[n*ld + k] (K contiguous).
DEVINL v16h load_b_frag(const _Float16* base, int ld) {
  const int lane = threadIdx.x & 31;
  const int n = lane & 15;
  const int g = lane >> 4;
  const _Float16* p = base + (size_t)n * ld + g * 8;
  v8h lo = *(const v8h*)p;
  v8h hi = *(const v8h*)(p + 16);
  v16h b;
#pragma unroll
  for (int i = 0; i < 8; ++i) { b[i] = lo[i]; b[i + 8] = hi[i]; }
  return b;
}

// ------------------------------------------------- async global -> LDS stage
DEVINL uint32_t lds_addr32(const void* p) { return (uint32_t)(uintptr_t)p; }

// Copy a 64 x 256-half row-major tile (ld = CN) into LDS (stride LDG halves)
// with 8 global_load_async_to_lds_b128 per thread (256 threads, 32KB total).
DEVINL void async_tile_load_64x256(const _Float16* gbase, uint32_t ldsbase) {
  const int tid = threadIdx.x;
#pragma unroll
  for (int i = 0; i < 8; ++i) {
    const int chunk = tid + i * 256;      // 0..2047 16B chunks
    const int row = chunk >> 5;           // 32 chunks per 512B row
    const int col = chunk & 31;
    const uint32_t loff = ldsbase + (uint32_t)(row * (LDG * 2) + col * 16);
    const void* g = (const void*)((const char*)gbase + (size_t)row * (CN * 2) + col * 16);
    asm volatile("global_load_async_to_lds_b128 %0, %1, off"
                 :: "v"(loff), "v"(g) : "memory");
  }
}
DEVINL void wait_async_all(void) {
  asm volatile("s_wait_asynccnt 0" ::: "memory");
}
DEVINL void wait_async_older8(void) {   // allow the 8 just-issued to stay in flight
  asm volatile("s_wait_asynccnt 8" ::: "memory");
}

// Reductions across a 16-lane half-wave (xor masks 8..1 stay inside group)
DEVINL float half_red_max(float v) {
#pragma unroll
  for (int m = 8; m; m >>= 1) v = fmaxf(v, __shfl_xor(v, m, 32));
  return v;
}
DEVINL float half_red_sum(float v) {
#pragma unroll
  for (int m = 8; m; m >>= 1) v += __shfl_xor(v, m, 32);
  return v;
}
DEVINL float full_red_sum(float v) {
#pragma unroll
  for (int m = 16; m; m >>= 1) v += __shfl_xor(v, m, 32);
  return v;
}

// ==================================================================
// Kernel 0: one-shot weight convert fp32 -> f16 (Q scaled by C^-0.5)
// ==================================================================
__global__ __launch_bounds__(256)
void wcvt_kernel(const float* __restrict__ Wq, const float* __restrict__ Wk,
                 const float* __restrict__ Wv, _Float16* __restrict__ Wh) {
  const int i = blockIdx.x * 256 + threadIdx.x;   // grid = 3*65536/256 = 768
  const int mat = i >> 16;
  const int off = i & 65535;
  const float* W = (mat == 0) ? Wq : (mat == 1) ? Wk : Wv;
  const float sc = (mat == 0) ? QSCALE : 1.0f;
  Wh[i] = (_Float16)(W[off] * sc);
}

// ==================================================================
// Kernel 1: LayerNorm + QKV projection (f16 out; Q pre-scaled, V transposed)
// grid = B*T/64 blocks of 256 threads (8 waves); block owns 64 tokens.
// ==================================================================
__global__ __launch_bounds__(256)
void ln_qkv_kernel(const float* __restrict__ x,
                   const float* __restrict__ gamma, const float* __restrict__ beta,
                   const _Float16* __restrict__ Wh,
                   const float* __restrict__ bq, const float* __restrict__ bk,
                   const float* __restrict__ bv,
                   _Float16* __restrict__ Qh, _Float16* __restrict__ Kh,
                   _Float16* __restrict__ Vt) {
  const int b    = blockIdx.x >> 6;
  const int t0   = (blockIdx.x & 63) * 64;
  const int lane = threadIdx.x & 31;
  const int wave = threadIdx.x >> 5;

  __shared__ __align__(16) _Float16 gn[64 * LDG];

  const float* xb = x + ((size_t)b * TN + t0) * CN;
  v4f g0  = *(const v4f*)(gamma + lane * 8);
  v4f g1  = *(const v4f*)(gamma + lane * 8 + 4);
  v4f be0 = *(const v4f*)(beta + lane * 8);
  v4f be1 = *(const v4f*)(beta + lane * 8 + 4);

  // --- LayerNorm: each wave normalizes one token per iteration ---
#pragma unroll
  for (int it = 0; it < 8; ++it) {
    const int tl = it * 8 + wave;
    const float* xr = xb + (size_t)tl * CN + lane * 8;
    v4f a0 = *(const v4f*)xr;
    v4f a1 = *(const v4f*)(xr + 4);
    float s = 0.f, q = 0.f;
#pragma unroll
    for (int j = 0; j < 4; ++j) {
      s += a0[j] + a1[j];
      q += a0[j] * a0[j] + a1[j] * a1[j];
    }
    s = full_red_sum(s);
    q = full_red_sum(q);
    const float mu   = s * (1.0f / CN);
    const float rstd = rsqrtf(q * (1.0f / CN) - mu * mu + LN_EPS);
    _Float16* gr = &gn[(size_t)tl * LDG + lane * 8];
#pragma unroll
    for (int j = 0; j < 4; ++j) {
      gr[j]     = (_Float16)((a0[j] - mu) * rstd * g0[j] + be0[j]);
      gr[j + 4] = (_Float16)((a1[j] - mu) * rstd * g1[j] + be1[j]);
    }
  }
  __syncthreads();

  // --- QKV GEMM: 48 N-tiles (3 matrices x 16 cout tiles), 6 per wave ---
  for (int idx = wave; idx < 48; idx += 8) {
    const int mat = idx >> 4;           // 0=Q 1=K 2=V
    const int ct  = idx & 15;
    const _Float16* W = Wh + (size_t)mat * CN * CN;
    const float* bia  = (mat == 0) ? bq : (mat == 1) ? bk : bv;

    v8f acc[4] = {};
    for (int kt = 0; kt < 8; ++kt) {
      v16h bb = load_b_frag(W + (size_t)(ct * 16) * CN + kt * 32, CN);
#pragma unroll
      for (int mt = 0; mt < 4; ++mt) {
        v16h a = load_a_frag(&gn[(size_t)(mt * 16) * LDG + kt * 32], LDG);
        acc[mt] = wmma_f16(a, bb, acc[mt]);
      }
    }
    const int   cl    = ct * 16 + (lane & 15);
    const float biasv = bia[cl] * ((mat == 0) ? QSCALE : 1.0f);
    if (mat == 2) {                     // V transposed: layout (b, c, t)
      _Float16* dst = Vt + ((size_t)b * CN + cl) * TN + t0;
#pragma unroll
      for (int mt = 0; mt < 4; ++mt)
#pragma unroll
        for (int r = 0; r < 8; ++r) {
          const int tl = mt * 16 + (lane >> 4) * 8 + r;
          dst[tl] = (_Float16)(acc[mt][r] + biasv);
        }
    } else {                            // Q/K row-major (b, t, c)
      _Float16* dst = ((mat == 0) ? Qh : Kh) + ((size_t)b * TN + t0) * CN + cl;
#pragma unroll
      for (int mt = 0; mt < 4; ++mt)
#pragma unroll
        for (int r = 0; r < 8; ++r) {
          const int tl = mt * 16 + (lane >> 4) * 8 + r;
          dst[(size_t)tl * CN] = (_Float16)(acc[mt][r] + biasv);
        }
    }
  }
}

// ==================================================================
// Kernel 2: softmax row statistics (max & 1/sum over s for each row t)
// grid = B*T/64 blocks; Q tile async-staged in LDS; waves split s-blocks.
// ==================================================================
__global__ __launch_bounds__(256)
void row_stats_kernel(const _Float16* __restrict__ Qh,
                      const _Float16* __restrict__ Kh,
                      float* __restrict__ rmax, float* __restrict__ rinv) {
  const int b    = blockIdx.x >> 6;
  const int t0   = (blockIdx.x & 63) * 64;
  const int lane = threadIdx.x & 31;
  const int wave = threadIdx.x >> 5;
  const _Float16* Qb = Qh + ((size_t)b * TN + t0) * CN;
  const _Float16* Kb = Kh + (size_t)b * TN * CN;

  __shared__ __align__(16) _Float16 Ql[64 * LDG];
  async_tile_load_64x256(Qb, lds_addr32(Ql));
  wait_async_all();
  __syncthreads();

  float rm[4][8], rs[4][8];
#pragma unroll
  for (int mt = 0; mt < 4; ++mt)
#pragma unroll
    for (int r = 0; r < 8; ++r) { rm[mt][r] = -3.0e38f; rs[mt][r] = 0.0f; }

  for (int sb = wave; sb < TN / 64; sb += 8) {
    const int s0 = sb * 64;
#pragma unroll
    for (int h = 0; h < 2; ++h) {        // two 32-wide s halves
      v8f acc[4][2] = {};
      for (int kt = 0; kt < 8; ++kt) {
        v16h b0 = load_b_frag(Kb + (size_t)(s0 + (h * 2 + 0) * 16) * CN + kt * 32, CN);
        v16h b1 = load_b_frag(Kb + (size_t)(s0 + (h * 2 + 1) * 16) * CN + kt * 32, CN);
#pragma unroll
        for (int mt = 0; mt < 4; ++mt) {
          v16h a = load_a_frag(&Ql[(size_t)(mt * 16) * LDG + kt * 32], LDG);
          acc[mt][0] = wmma_f16(a, b0, acc[mt][0]);
          acc[mt][1] = wmma_f16(a, b1, acc[mt][1]);
        }
      }
      // online softmax update (rows live in lane half-groups)
#pragma unroll
      for (int mt = 0; mt < 4; ++mt)
#pragma unroll
        for (int r = 0; r < 8; ++r) {
          float v = fmaxf(acc[mt][0][r], acc[mt][1][r]);
          float tmax = half_red_max(v);
          float nm = fmaxf(rm[mt][r], tmax);
          float es = __expf(acc[mt][0][r] - nm) + __expf(acc[mt][1][r] - nm);
          es = half_red_sum(es);
          rs[mt][r] = rs[mt][r] * __expf(rm[mt][r] - nm) + es;
          rm[mt][r] = nm;
        }
    }
  }

  // cross-wave merge
  __shared__ float smx[8][64];
  __shared__ float ssm[8][64];
  if ((lane & 15) == 0) {
    const int g = lane >> 4;
#pragma unroll
    for (int mt = 0; mt < 4; ++mt)
#pragma unroll
      for (int r = 0; r < 8; ++r) {
        smx[wave][mt * 16 + g * 8 + r] = rm[mt][r];
        ssm[wave][mt * 16 + g * 8 + r] = rs[mt][r];
      }
  }
  __syncthreads();
  const int tid = threadIdx.x;
  if (tid < 64) {
    float m = -3.0e38f;
#pragma unroll
    for (int w = 0; w < 8; ++w) m = fmaxf(m, smx[w][tid]);
    float s = 0.0f;
#pragma unroll
    for (int w = 0; w < 8; ++w) s += ssm[w][tid] * __expf(smx[w][tid] - m);
    rmax[(size_t)b * TN + t0 + tid] = m;
    rinv[(size_t)b * TN + t0 + tid] = 1.0f / s;
  }
}

// ==================================================================
// Kernel 3: out[b,s,c] = x[b,s,c] + sum_t P[t,s] * V[t,c]   (P^T · V)
// K tile LDS-resident; Q tiles double-buffered via async LDS loads.
// ==================================================================
__global__ __launch_bounds__(256)
void attn_out_kernel(const _Float16* __restrict__ Qh,
                     const _Float16* __restrict__ Kh,
                     const _Float16* __restrict__ Vt,
                     const float* __restrict__ rmax, const float* __restrict__ rinv,
                     const float* __restrict__ x, float* __restrict__ out) {
  const int b    = blockIdx.x >> 6;
  const int s0   = (blockIdx.x & 63) * 64;
  const int lane = threadIdx.x & 31;
  const int wave = threadIdx.x >> 5;
  const int tid  = threadIdx.x;
  const _Float16* Qb = Qh + (size_t)b * TN * CN;
  const _Float16* Kb = Kh + (size_t)b * TN * CN;
  const _Float16* Vb = Vt + (size_t)b * CN * TN;
  const int c0 = wave * 32;

  __shared__ __align__(16) _Float16 Kl[64 * LDG];      // K s-tile, resident
  __shared__ __align__(16) _Float16 Qd[2][64 * LDG];   // double-buffered Q
  __shared__ __align__(16) _Float16 pT[64 * LDP];      // P^T: [s_local][t_local]
  __shared__ float sm[64];
  __shared__ float si[64];

  async_tile_load_64x256(Kb + (size_t)s0 * CN, lds_addr32(Kl));
  async_tile_load_64x256(Qb, lds_addr32(Qd[0]));
  wait_async_all();
  __syncthreads();

  v8f oacc[4][2] = {};
  const int mtw = wave >> 1;            // P tile pair: same mt, nt in {nt0,nt0+1}
  const int nt0 = (wave & 1) * 2;

  for (int tb = 0; tb < TN / 64; ++tb) {
    const int cur = tb & 1;
    const int tg  = tb * 64;
    __syncthreads();  // (A) prior pT readers + prior Qd[1-cur] readers done
    const bool prefetch = (tb + 1 < TN / 64);
    if (prefetch)
      async_tile_load_64x256(Qb + (size_t)(tg + 64) * CN, lds_addr32(Qd[1 - cur]));
    if (tid < 64) {
      sm[tid] = rmax[(size_t)b * TN + tg + tid];
      si[tid] = rinv[(size_t)b * TN + tg + tid];
    }
    // Wait so that all chunks of Qd[cur] have landed. Async loads complete in
    // order, so <=8 leaves only the 8 just-issued prefetch chunks in flight.
    // On the last iteration nothing new was issued -> must drain fully.
    if (prefetch) wait_async_older8();
    else          wait_async_all();
    __syncthreads();   // (B) all threads' Qd[cur] writes + stats visible

    // --- score tiles -> normalized P^T (2 tiles per wave, shared A-frag) ---
    {
      v8f acc0 = {}, acc1 = {};
      for (int kt = 0; kt < 8; ++kt) {
        v16h a  = load_a_frag(&Qd[cur][(size_t)(mtw * 16) * LDG + kt * 32], LDG);
        v16h b0 = load_b_frag(&Kl[(size_t)((nt0 + 0) * 16) * LDG + kt * 32], LDG);
        v16h b1 = load_b_frag(&Kl[(size_t)((nt0 + 1) * 16) * LDG + kt * 32], LDG);
        acc0 = wmma_f16(a, b0, acc0);
        acc1 = wmma_f16(a, b1, acc1);
      }
      const int nlo = (nt0 + 0) * 16 + (lane & 15);
      const int nhi = (nt0 + 1) * 16 + (lane & 15);
      const int g8 = (lane >> 4) * 8;
#pragma unroll
      for (int r = 0; r < 8; ++r) {
        const int tl = mtw * 16 + g8 + r;
        pT[(size_t)nlo * LDP + tl] = (_Float16)(__expf(acc0[r] - sm[tl]) * si[tl]);
        pT[(size_t)nhi * LDP + tl] = (_Float16)(__expf(acc1[r] - sm[tl]) * si[tl]);
      }
    }
    __syncthreads();   // (C) pT ready

    // --- O += P^T (64s x 64t) · V (64t x 32c per wave) ---
#pragma unroll
    for (int kt = 0; kt < 2; ++kt) {
      v16h b0 = load_b_frag(Vb + (size_t)(c0 + 0)  * TN + tg + kt * 32, TN);
      v16h b1 = load_b_frag(Vb + (size_t)(c0 + 16) * TN + tg + kt * 32, TN);
#pragma unroll
      for (int mts = 0; mts < 4; ++mts) {
        v16h a = load_a_frag(&pT[(size_t)(mts * 16) * LDP + kt * 32], LDP);
        oacc[mts][0] = wmma_f16(a, b0, oacc[mts][0]);
        oacc[mts][1] = wmma_f16(a, b1, oacc[mts][1]);
      }
    }
  }

  // --- epilogue: residual add, fp32 store ---
#pragma unroll
  for (int mts = 0; mts < 4; ++mts)
#pragma unroll
    for (int nc = 0; nc < 2; ++nc) {
      const int c = c0 + nc * 16 + (lane & 15);
#pragma unroll
      for (int r = 0; r < 8; ++r) {
        const int s = s0 + mts * 16 + (lane >> 4) * 8 + r;
        const size_t idx = ((size_t)b * TN + s) * CN + c;
        out[idx] = x[idx] + oacc[mts][nc][r];
      }
    }
}

// ==================================================================
extern "C" void kernel_launch(void* const* d_in, const int* in_sizes, int n_in,
                              void* d_out, int out_size, void* d_ws, size_t ws_size,
                              hipStream_t stream) {
  (void)in_sizes; (void)n_in; (void)out_size; (void)ws_size;
  const float* x  = (const float*)d_in[0];
  const float* g  = (const float*)d_in[1];
  const float* be = (const float*)d_in[2];
  const float* Wq = (const float*)d_in[3];
  const float* bq = (const float*)d_in[4];
  const float* Wk = (const float*)d_in[5];
  const float* bk = (const float*)d_in[6];
  const float* Wv = (const float*)d_in[7];
  const float* bv = (const float*)d_in[8];
  float* out = (float*)d_out;

  char* w = (char*)d_ws;
  _Float16* Qh = (_Float16*)w; w += (size_t)BN * TN * CN * sizeof(_Float16);
  _Float16* Kh = (_Float16*)w; w += (size_t)BN * TN * CN * sizeof(_Float16);
  _Float16* Vt = (_Float16*)w; w += (size_t)BN * TN * CN * sizeof(_Float16);
  float* rmaxp = (float*)w;    w += (size_t)BN * TN * sizeof(float);
  float* rinvp = (float*)w;    w += (size_t)BN * TN * sizeof(float);
  _Float16* Wh = (_Float16*)w; // 3 * 256*256 f16 = 384KB

  dim3 blk(256);
  wcvt_kernel<<<dim3(3 * CN * CN / 256), blk, 0, stream>>>(Wq, Wk, Wv, Wh);
  ln_qkv_kernel<<<dim3(BN * TN / 64), blk, 0, stream>>>(x, g, be, Wh, bq, bk, bv,
                                                        Qh, Kh, Vt);
  row_stats_kernel<<<dim3(BN * TN / 64), blk, 0, stream>>>(Qh, Kh, rmaxp, rinvp);
  attn_out_kernel<<<dim3(BN * TN / 64), blk, 0, stream>>>(Qh, Kh, Vt, rmaxp, rinvp,
                                                          x, out);
}
